// MultiHeadSelfAttention_55654186222044
// MI455X (gfx1250) — matrix operand order repaired and checked by reading, not compiled, not run on hardware
//
#include <hip/hip_runtime.h>

typedef __attribute__((ext_vector_type(8)))  float  v8f;
typedef __attribute__((ext_vector_type(8)))  __bf16 v8bf;
typedef __attribute__((ext_vector_type(16))) __bf16 v16bf;

#define WMMA_BF16(a, b, c) \
  __builtin_amdgcn_wmma_f32_16x16x32_bf16(false, (a), false, (b), (short)0, (c), false, false)

// Problem constants
static constexpr int Bsz = 2, T = 1024, E = 1024, H = 16, D = 64;
static constexpr int M = Bsz * H * T;  // 32768 "token" rows (b,h,t)

// ---------------------------------------------------------------------------
// fp32 -> bf16 elementwise convert (weights)
// ---------------------------------------------------------------------------
__global__ __launch_bounds__(256) void cvt_f32_bf16(const float* __restrict__ src,
                                                    __bf16* __restrict__ dst, int n) {
  int i = blockIdx.x * 256 + threadIdx.x;
  if (i < n) dst[i] = (__bf16)src[i];
}

// ---------------------------------------------------------------------------
// Kernel 1: q/k/v = X(32768x64) @ Wflat(1024x64)^T + bias   (K = 64)
// One wave: 16 rows x 64 cols for q, k and v  (12 tiles, 24 WMMAs)
// ---------------------------------------------------------------------------
__global__ __launch_bounds__(256) void qkv_kernel(
    const float* __restrict__ x,
    const __bf16* __restrict__ wq, const __bf16* __restrict__ wk, const __bf16* __restrict__ wv,
    const float* __restrict__ bq, const float* __restrict__ bk, const float* __restrict__ bv,
    __bf16* __restrict__ q, __bf16* __restrict__ k, __bf16* __restrict__ v) {
  const int wid  = blockIdx.x * 8 + (threadIdx.x >> 5);
  const int lane = threadIdx.x & 31;
  const int mt = wid >> 4;        // 0..2047  (16-row M tile)
  const int ng = wid & 15;        // 0..15    (64-col N group)
  const int l15 = lane & 15, hi = lane >> 4;

  // A-operand row: m=(b,h,t) -> x[b][t][h*64+d]
  const int m  = mt * 16 + l15;
  const int bh = m >> 10, t = m & 1023;
  const int bb = bh >> 4, hh = bh & 15;
  const float* xrow = x + ((size_t)(bb * 1024 + t)) * 1024 + hh * 64;

  v8f cq[4], ck[4], cv[4];
#pragma unroll
  for (int nt = 0; nt < 4; ++nt) {
    const int n = ng * 64 + nt * 16 + l15;
    const float q0 = bq[n], k0 = bk[n], v0 = bv[n];
#pragma unroll
    for (int r = 0; r < 8; ++r) { cq[nt][r] = q0; ck[nt][r] = k0; cv[nt][r] = v0; }
  }

  const int kb8 = hi * 8, kb16 = hi * 16;
#pragma unroll
  for (int s = 0; s < 64; s += 32) {
    // x is read exactly once by this pipeline: non-temporal loads keep L2 for
    // the reused q/k/v/cat/Wf working set.
    v8f alo, ahi;
#pragma unroll
    for (int j = 0; j < 8; ++j) {
      alo[j] = __builtin_nontemporal_load(xrow + s + kb16 + j);
      ahi[j] = __builtin_nontemporal_load(xrow + s + kb16 + 8 + j);
    }
    v16bf a;
#pragma unroll
    for (int j = 0; j < 8; ++j) { a[j] = (__bf16)alo[j]; a[8 + j] = (__bf16)ahi[j]; }
#pragma unroll
    for (int nt = 0; nt < 4; ++nt) {
      const int n = ng * 64 + nt * 16 + l15;
      cq[nt] = WMMA_BF16(a, *(const v16bf*)(wq + (size_t)n * 64 + s + kb16), cq[nt]);
      ck[nt] = WMMA_BF16(a, *(const v16bf*)(wk + (size_t)n * 64 + s + kb16), ck[nt]);
      cv[nt] = WMMA_BF16(a, *(const v16bf*)(wv + (size_t)n * 64 + s + kb16), cv[nt]);
    }
  }
#pragma unroll
  for (int nt = 0; nt < 4; ++nt) {
    const int n = ng * 64 + nt * 16 + l15;
#pragma unroll
    for (int r = 0; r < 8; ++r) {
      const size_t off = (size_t)(mt * 16 + r + 8 * hi) * 1024 + n;
      q[off] = (__bf16)cq[nt][r];
      k[off] = (__bf16)ck[nt][r];
      v[off] = (__bf16)cv[nt][r];
    }
  }
}

// ---------------------------------------------------------------------------
// Kernel 2: per (b,h,i) combo: S(64x64) = Q_i^T K_i / 8 over T=1024, then
// row-softmax -> wsm (bf16). Q/K chunks staged transposed through LDS.
// Block = 512 threads (16 waves), one 16x16 S-tile per wave.
// ---------------------------------------------------------------------------
__global__ __launch_bounds__(512) void scores_softmax_kernel(
    const __bf16* __restrict__ q, const __bf16* __restrict__ k, __bf16* __restrict__ wsm) {
  __shared__ __bf16 Qt[64 * 32];   // [p][t_in_chunk]
  __shared__ __bf16 Kt[64 * 32];   // [qcol][t_in_chunk]
  __shared__ float  S[64 * 64];

  const int combo = blockIdx.x;           // (b,h,i), 512 total
  const int bh = combo >> 4, ii = combo & 15;
  const int tid = threadIdx.x;
  const int lane = tid & 31, w = tid >> 5;
  const int l15 = lane & 15, hi = lane >> 4;
  const int tp = (w >> 2) * 16, tq = (w & 3) * 16;
  const size_t base = (size_t)bh * 1024 * 1024 + (size_t)ii * 64;

  v8f c = {};
  for (int chunk = 0; chunk < 32; ++chunk) {
    const int t0 = chunk * 32;
    {  // cooperative transpose-load: threads 0..255 -> Q, 256..511 -> K
      const __bf16* src = (tid < 256) ? q : k;
      __bf16* dst = (tid < 256) ? Qt : Kt;
      const int id = tid & 255;
      const int tt = id >> 3;              // t within chunk: 0..31
      const int pg = (id & 7) * 8;         // 8-wide col group
      v8bf val = *(const v8bf*)(src + base + (size_t)(t0 + tt) * 1024 + pg);
#pragma unroll
      for (int j = 0; j < 8; ++j) dst[(pg + j) * 32 + tt] = val[j];
    }
    __syncthreads();
    v16bf a, b;
    {
      const __bf16* ap = &Qt[(tp + l15) * 32 + hi * 16];
      v8bf alo = *(const v8bf*)ap;
      v8bf ahi2 = *(const v8bf*)(ap + 8);
#pragma unroll
      for (int j = 0; j < 8; ++j) { a[j] = alo[j]; a[8 + j] = ahi2[j]; }
      b = *(const v16bf*)&Kt[(tq + l15) * 32 + hi * 16];
    }
    c = WMMA_BF16(a, b, c);
    __syncthreads();
  }
#pragma unroll
  for (int r = 0; r < 8; ++r)
    S[(tp + r + 8 * hi) * 64 + tq + l15] = c[r] * 0.125f;  // / sqrt(64)
  __syncthreads();

  if (tid < 64) {  // softmax over last axis, one row per thread
    float mx = -3.4e38f;
    for (int j = 0; j < 64; ++j) mx = fmaxf(mx, S[tid * 64 + j]);
    float sum = 0.f;
    for (int j = 0; j < 64; ++j) sum += __expf(S[tid * 64 + j] - mx);
    const float inv = 1.f / sum;
    for (int j = 0; j < 64; ++j)
      wsm[(size_t)combo * 4096 + tid * 64 + j] = (__bf16)(__expf(S[tid * 64 + j] - mx) * inv);
  }
}

// ---------------------------------------------------------------------------
// Kernel 3: per combo: O_i(1024x64) = V_i @ W^T  (K = 64) -> cat(32768x1024)
// ---------------------------------------------------------------------------
__global__ __launch_bounds__(256) void av_kernel(
    const __bf16* __restrict__ v, const __bf16* __restrict__ wsm, __bf16* __restrict__ cat) {
  const int wid  = blockIdx.x * 8 + (threadIdx.x >> 5);
  const int lane = threadIdx.x & 31;
  const int combo = wid >> 6;   // 0..511
  const int mt    = wid & 63;   // 16-row t tile
  const int bh = combo >> 4, ii = combo & 15;
  const int l15 = lane & 15, hi = lane >> 4;
  const size_t vrow = (size_t)(bh * 1024 + mt * 16 + l15) * 1024 + (size_t)ii * 64;

  v8f c[4] = {};
#pragma unroll
  for (int s = 0; s < 64; s += 32) {
    v8bf alo = *(const v8bf*)(v + vrow + s + hi * 16);
    v8bf ahi = *(const v8bf*)(v + vrow + s + 8 + hi * 16);
    v16bf a;
#pragma unroll
    for (int j = 0; j < 8; ++j) { a[j] = alo[j]; a[8 + j] = ahi[j]; }
#pragma unroll
    for (int nt = 0; nt < 4; ++nt) {
      const int n = nt * 16 + l15;
      v16bf b = *(const v16bf*)(wsm + (size_t)combo * 4096 + (size_t)n * 64 + s + hi * 16);
      c[nt] = WMMA_BF16(a, b, c[nt]);
    }
  }
#pragma unroll
  for (int nt = 0; nt < 4; ++nt)
#pragma unroll
    for (int r = 0; r < 8; ++r) {
      const int row = bh * 1024 + mt * 16 + r + 8 * hi;
      cat[(size_t)row * 1024 + ii * 64 + nt * 16 + l15] = (__bf16)c[nt][r];
    }
}

// ---------------------------------------------------------------------------
// Kernel 4 (dominant, 68.7 GFLOP): out = cat(32768x1024) @ Wf^T + bf  (fp32)
// One wave: 32x64 strip (2 M-subtiles x 4 N-tiles). K loop 1024:
// 32 steps x 8 WMMAs with B reused across both M-subtiles (8 WMMA : 8 loads).
// Output stores are non-temporal: write-once 128MB stream must not evict the
// re-used cat (64MB) / Wf (2MB) working set from the 192MB L2.
// ---------------------------------------------------------------------------
__global__ __launch_bounds__(256) void final_kernel(
    const __bf16* __restrict__ cat, const __bf16* __restrict__ wf,
    const float* __restrict__ bfv, float* __restrict__ out) {
  const int wid  = blockIdx.x * 8 + (threadIdx.x >> 5);
  const int lane = threadIdx.x & 31;
  const int mt2 = wid >> 4;   // 0..1023  (32-row M tile)
  const int ng  = wid & 15;   // 0..15    (64-col N group)
  const int l15 = lane & 15, hi = lane >> 4;

  const __bf16* arow0 = cat + (size_t)(mt2 * 32 + l15) * 1024;       // M-subtile 0
  const __bf16* arow1 = arow0 + (size_t)16 * 1024;                   // M-subtile 1

  v8f c0[4], c1[4];
#pragma unroll
  for (int nt = 0; nt < 4; ++nt) {
    const float b0 = bfv[ng * 64 + nt * 16 + l15];
#pragma unroll
    for (int r = 0; r < 8; ++r) { c0[nt][r] = b0; c1[nt][r] = b0; }
  }

  for (int s = 0; s < 1024; s += 32) {
    __builtin_prefetch(arow0 + s + 64, 0, 0);  // global_prefetch_b8 next A chunk
    __builtin_prefetch(arow1 + s + 64, 0, 0);
    v8bf a0lo = *(const v8bf*)(arow0 + s + hi * 16);
    v8bf a0hi = *(const v8bf*)(arow0 + s + 8 + hi * 16);
    v8bf a1lo = *(const v8bf*)(arow1 + s + hi * 16);
    v8bf a1hi = *(const v8bf*)(arow1 + s + 8 + hi * 16);
    v16bf a0, a1;
#pragma unroll
    for (int j = 0; j < 8; ++j) {
      a0[j] = a0lo[j]; a0[8 + j] = a0hi[j];
      a1[j] = a1lo[j]; a1[8 + j] = a1hi[j];
    }
#pragma unroll
    for (int nt = 0; nt < 4; ++nt) {
      const int n = ng * 64 + nt * 16 + l15;
      v16bf b = *(const v16bf*)(wf + (size_t)n * 1024 + s + hi * 16);
      c0[nt] = WMMA_BF16(a0, b, c0[nt]);   // B reused by both M-subtiles
      c1[nt] = WMMA_BF16(a1, b, c1[nt]);
    }
  }
#pragma unroll
  for (int nt = 0; nt < 4; ++nt) {
    const int n = ng * 64 + nt * 16 + l15;
#pragma unroll
    for (int r = 0; r < 8; ++r) {
      const int row0 = mt2 * 32 + r + 8 * hi;
      __builtin_nontemporal_store(c0[nt][r], out + (size_t)row0 * 1024 + n);
      __builtin_nontemporal_store(c1[nt][r], out + (size_t)(row0 + 16) * 1024 + n);
    }
  }
}

// ---------------------------------------------------------------------------
// Host-side launcher
// ---------------------------------------------------------------------------
extern "C" void kernel_launch(void* const* d_in, const int* in_sizes, int n_in,
                              void* d_out, int out_size, void* d_ws, size_t ws_size,
                              hipStream_t stream) {
  const float* x  = (const float*)d_in[0];
  const float* Wq = (const float*)d_in[1];
  const float* bq = (const float*)d_in[2];
  const float* Wk = (const float*)d_in[3];
  const float* bk = (const float*)d_in[4];
  const float* Wv = (const float*)d_in[5];
  const float* bv = (const float*)d_in[6];
  const float* Wf = (const float*)d_in[7];
  const float* bf = (const float*)d_in[8];
  float* out = (float*)d_out;

  // workspace layout (bytes, all offsets 128KB-aligned)
  char* ws = (char*)d_ws;
  const size_t W_SMALL = (size_t)H * D * D * sizeof(__bf16);   // 128 KB each
  const size_t W_BIG   = (size_t)E * E * sizeof(__bf16);       // 2 MB
  const size_t QKV     = (size_t)M * E * sizeof(__bf16);       // 64 MB each
  const size_t WSM     = (size_t)Bsz * H * H * D * D * sizeof(__bf16);  // 4 MB

  __bf16* wq_b  = (__bf16*)(ws);
  __bf16* wk_b  = (__bf16*)(ws + W_SMALL);
  __bf16* wv_b  = (__bf16*)(ws + 2 * W_SMALL);
  __bf16* wf_b  = (__bf16*)(ws + 3 * W_SMALL);
  __bf16* qb    = (__bf16*)(ws + 3 * W_SMALL + W_BIG);
  __bf16* kb    = (__bf16*)(ws + 3 * W_SMALL + W_BIG + QKV);
  __bf16* vb    = (__bf16*)(ws + 3 * W_SMALL + W_BIG + 2 * QKV);
  __bf16* wsm_b = (__bf16*)(ws + 3 * W_SMALL + W_BIG + 3 * QKV);
  __bf16* cat_b = (__bf16*)(ws + 3 * W_SMALL + W_BIG + 3 * QKV + WSM);

  // 1) convert weights fp32 -> bf16
  const int nsmall = H * D * D;            // 65536
  const int nbig   = E * E;                // 1048576
  cvt_f32_bf16<<<(nsmall + 255) / 256, 256, 0, stream>>>(Wq, wq_b, nsmall);
  cvt_f32_bf16<<<(nsmall + 255) / 256, 256, 0, stream>>>(Wk, wk_b, nsmall);
  cvt_f32_bf16<<<(nsmall + 255) / 256, 256, 0, stream>>>(Wv, wv_b, nsmall);
  cvt_f32_bf16<<<(nbig + 255) / 256, 256, 0, stream>>>(Wf, wf_b, nbig);

  // 2) q/k/v projections: 32768 waves (2048 M-tiles x 16 N-groups), 8 waves/block
  qkv_kernel<<<4096, 256, 0, stream>>>(x, wq_b, wk_b, wv_b, bq, bk, bv, qb, kb, vb);

  // 3) feature-attention scores + softmax: one block per (b,h,i) combo
  scores_softmax_kernel<<<Bsz * H * H, 512, 0, stream>>>(qb, kb, wsm_b);

  // 4) o = V @ W^T -> cat: 32768 waves (512 combos x 64 t-tiles)
  av_kernel<<<4096, 256, 0, stream>>>(vb, wsm_b, cat_b);

  // 5) final Linear(E,E): 16384 waves (1024 32-row M-tiles x 16 N-groups)
  final_kernel<<<2048, 256, 0, stream>>>(cat_b, wf_b, bf, out);
}
